// TopLineEncoder_30846455119979
// MI455X (gfx1250) — compile-verified, matrix operand order checked by reading, not run on hardware
//
#include <hip/hip_runtime.h>

// Problem constants (match reference)
#define B_ 4096
#define L_ 80
#define V_ 256
#define H_ 512

// GEMM tiling
#define BM 128
#define BN 128
#define BK 32
#define SK 40                        // padded LDS row stride (bf16 elems): 80B, 16B aligned
#define TILE_LDS_B (BM * SK * 2)     // 10240 bytes per tile buffer
#define KSTEPS (H_ / BK)             // 16

typedef __attribute__((ext_vector_type(16))) __bf16       bf16x16;
typedef __attribute__((ext_vector_type(8)))  float        f32x8;
typedef __attribute__((ext_vector_type(4)))  float        f32x4;
typedef __attribute__((ext_vector_type(4)))  unsigned int u32x4;
typedef __attribute__((ext_vector_type(2)))  unsigned int u32x2;

// f32 -> bf16 with round-to-nearest-even (bit pattern as ushort)
static __device__ __forceinline__ unsigned short f32_to_bf16(float f) {
    unsigned u = __builtin_bit_cast(unsigned, f);
    u += 0x7FFFu + ((u >> 16) & 1u);
    return (unsigned short)(u >> 16);
}

// 16 x bf16 WMMA fragment slice: two 16B loads; K 0..7 at p, K 16..23 at p+16
// (ISA 16-bit A/B layout, lane<16; lanes>=16 shift p by 8 elements).
static __device__ __forceinline__ bf16x16 load_frag32(const unsigned short* p) {
    union { bf16x16 v; u32x4 q[2]; } u;
    u.q[0] = *(const u32x4*)(p);
    u.q[1] = *(const u32x4*)(p + 16);
    return u.v;
}

// Async global->LDS 16B copy (GVS mode: SGPR64 base + VGPR32 byte offset).
// VDST VGPR carries the LDS byte address. Tracked by ASYNCcnt.
static __device__ __forceinline__ void async_load_b128(unsigned lds_byte,
                                                       const void* sbase,
                                                       unsigned gbyte_off) {
    asm volatile("global_load_async_to_lds_b128 %0, %1, %2"
                 :
                 : "v"(lds_byte), "v"(gbyte_off), "s"(sbase)
                 : "memory");
}

static __device__ __forceinline__ void wait_async0() {
#if __has_builtin(__builtin_amdgcn_s_wait_asynccnt)
    __builtin_amdgcn_s_wait_asynccnt(0);
#else
    asm volatile("s_wait_asynccnt 0x0" ::: "memory");
#endif
}

// ---------------------------------------------------------------------------
// Kernel 1: W2 [512][512] f32 row-major  ->  w2t [n][k] bf16 (K-contiguous)
// ---------------------------------------------------------------------------
__global__ void w2_transpose_bf16(const float* __restrict__ W2,
                                  unsigned short* __restrict__ w2t) {
    const int id = blockIdx.x * 256 + threadIdx.x;   // 0 .. 512*512-1
    const int k = id >> 9;
    const int n = id & (H_ - 1);
    w2t[n * H_ + k] = f32_to_bf16(W2[id]);
}

// ---------------------------------------------------------------------------
// Kernel 2: gather-sum over 80 rows of W1, + b1, ReLU, emit bf16 h[B][512]
// One wave32 per batch row; lane j owns columns {j*4+c + k*128 : k<4, c<4}.
// ---------------------------------------------------------------------------
__global__ void gather_bias_relu_bf16(const int* __restrict__ msg,
                                      const float* __restrict__ W1,
                                      const float* __restrict__ b1,
                                      unsigned short* __restrict__ hb) {
    const int lane = threadIdx.x & 31;
    const int wave = threadIdx.x >> 5;
    const int row  = blockIdx.x * 8 + wave;

    const int* __restrict__ mrow = msg + row * L_;

    f32x4 acc0 = {0.f, 0.f, 0.f, 0.f};
    f32x4 acc1 = {0.f, 0.f, 0.f, 0.f};
    f32x4 acc2 = {0.f, 0.f, 0.f, 0.f};
    f32x4 acc3 = {0.f, 0.f, 0.f, 0.f};

    int c = mrow[0];
    for (int l = 0; l < L_; ++l) {
        const float* __restrict__ wr =
            W1 + (size_t)(l * V_ + c) * H_ + lane * 4;
        int cn = 0;
        if (l + 1 < L_) {
            cn = mrow[l + 1];
            // prefetch next gathered row (global_prefetch_b8)
            __builtin_prefetch(W1 + (size_t)((l + 1) * V_ + cn) * H_ + lane * 4, 0, 3);
        }
        acc0 += *(const f32x4*)(wr + 0);
        acc1 += *(const f32x4*)(wr + 128);
        acc2 += *(const f32x4*)(wr + 256);
        acc3 += *(const f32x4*)(wr + 384);
        c = cn;
    }

    f32x4 accs[4] = {acc0, acc1, acc2, acc3};
    unsigned short* __restrict__ hrow = hb + (size_t)row * H_;
#pragma unroll
    for (int k = 0; k < 4; ++k) {
        const int colbase = k * 128 + lane * 4;
        f32x4 v = accs[k] + *(const f32x4*)(b1 + colbase);
        v.x = fmaxf(v.x, 0.f);
        v.y = fmaxf(v.y, 0.f);
        v.z = fmaxf(v.z, 0.f);
        v.w = fmaxf(v.w, 0.f);
        u32x2 pk;
        pk.x = (unsigned)f32_to_bf16(v.x) | ((unsigned)f32_to_bf16(v.y) << 16);
        pk.y = (unsigned)f32_to_bf16(v.z) | ((unsigned)f32_to_bf16(v.w) << 16);
        *(u32x2*)(hrow + colbase) = pk;
    }
}

// ---------------------------------------------------------------------------
// Kernel 3: out = relu(h @ W2 + b2), LDS double-buffered WMMA GEMM.
// Block = 256 threads (8 waves, 2 m-waves x 4 n-waves) -> 128x128 tile.
// K staged in 32-wide slabs via global_load_async_to_lds_b128 (ASYNCcnt).
// ---------------------------------------------------------------------------
__global__ void __launch_bounds__(256)
gemm_relu_wmma(const unsigned short* __restrict__ hb,
               const unsigned short* __restrict__ w2t,
               const float* __restrict__ b2,
               float* __restrict__ out) {
    __shared__ unsigned short As[2][BM * SK];
    __shared__ unsigned short Bs[2][BN * SK];

    const int tid  = threadIdx.x;
    const int lane = tid & 31;
    const int wave = tid >> 5;
    const int mw   = wave & 1;   // 0..1 : 64-row half
    const int nw   = wave >> 1;  // 0..3 : 32-col slice
    const int lm   = lane & 15;
    const int khe  = (lane >> 4) * 8;   // K-half select, in elements

    // staging geometry: 128 rows x 32 elems = 512 x 16B jobs, 2 per thread
    const int r0 = tid >> 2,       s0 = tid & 3;
    const int r1 = (tid + 256) >> 2, s1 = (tid + 256) & 3;

    const unsigned as_base = (unsigned)(size_t)(void*)&As[0][0];
    const unsigned bs_base = (unsigned)(size_t)(void*)&Bs[0][0];
    const unsigned short* ga = hb  + (size_t)blockIdx.x * BM * H_;
    const unsigned short* gb = w2t + (size_t)blockIdx.y * BN * H_;

    const unsigned gvo0 = (unsigned)(r0 * (H_ * 2) + s0 * 16);  // bytes
    const unsigned gvo1 = (unsigned)(r1 * (H_ * 2) + s1 * 16);
    const unsigned lvo0 = (unsigned)(r0 * (SK * 2) + s0 * 16);
    const unsigned lvo1 = (unsigned)(r1 * (SK * 2) + s1 * 16);

    // prologue: stage slab 0 into buffer 0
    async_load_b128(as_base + lvo0, ga, gvo0);
    async_load_b128(bs_base + lvo0, gb, gvo0);
    async_load_b128(as_base + lvo1, ga, gvo1);
    async_load_b128(bs_base + lvo1, gb, gvo1);

    f32x8 acc[4][2];
#pragma unroll
    for (int t = 0; t < 4; ++t)
#pragma unroll
        for (int u = 0; u < 2; ++u)
            acc[t][u] = (f32x8){0.f, 0.f, 0.f, 0.f, 0.f, 0.f, 0.f, 0.f};

    for (int ki = 0; ki < KSTEPS; ++ki) {
        const int buf = ki & 1;
        wait_async0();        // this wave's portion of slab ki landed in LDS
        __syncthreads();      // everyone's portion landed; prior compute on buf^1 done

        if (ki + 1 < KSTEPS) {
            const unsigned gk = (unsigned)((ki + 1) * BK * 2);
            const unsigned lb = (unsigned)((buf ^ 1) * TILE_LDS_B);
            async_load_b128(as_base + lb + lvo0, ga, gvo0 + gk);
            async_load_b128(bs_base + lb + lvo0, gb, gvo0 + gk);
            async_load_b128(as_base + lb + lvo1, ga, gvo1 + gk);
            async_load_b128(bs_base + lb + lvo1, gb, gvo1 + gk);
        }

        const unsigned short* Ab = &As[buf][0];
        const unsigned short* Bb = &Bs[buf][0];

        bf16x16 bfr[2];
#pragma unroll
        for (int u = 0; u < 2; ++u)
            bfr[u] = load_frag32(Bb + (nw * 32 + u * 16 + lm) * SK + khe);
#pragma unroll
        for (int t = 0; t < 4; ++t) {
            bf16x16 a = load_frag32(Ab + (mw * 64 + t * 16 + lm) * SK + khe);
            acc[t][0] = __builtin_amdgcn_wmma_f32_16x16x32_bf16(
                false, a, false, bfr[0], (short)0, acc[t][0], false, false);
            acc[t][1] = __builtin_amdgcn_wmma_f32_16x16x32_bf16(
                false, a, false, bfr[1], (short)0, acc[t][1], false, false);
        }
    }

    // C/D layout: VGPR r -> M = r + 8*(lane>=16), N = lane%16
    const int rbase = blockIdx.x * BM + mw * 64 + (lane >> 4) * 8;
    const int cbase = blockIdx.y * BN + nw * 32;
#pragma unroll
    for (int t = 0; t < 4; ++t) {
#pragma unroll
        for (int u = 0; u < 2; ++u) {
            const int col  = cbase + u * 16 + lm;
            const float bias = b2[col];
#pragma unroll
            for (int r = 0; r < 8; ++r) {
                out[(size_t)(rbase + t * 16 + r) * H_ + col] =
                    fmaxf(acc[t][u][r] + bias, 0.f);
            }
        }
    }
}

// ---------------------------------------------------------------------------
extern "C" void kernel_launch(void* const* d_in, const int* in_sizes, int n_in,
                              void* d_out, int out_size, void* d_ws, size_t ws_size,
                              hipStream_t stream) {
    (void)in_sizes; (void)n_in; (void)out_size; (void)ws_size;

    const int*   msg = (const int*)d_in[0];
    const float* W1  = (const float*)d_in[1];
    const float* b1  = (const float*)d_in[2];
    const float* W2  = (const float*)d_in[3];
    const float* b2  = (const float*)d_in[4];
    float* out = (float*)d_out;

    // workspace carve: h (bf16, 4 MB) | W2^T (bf16, 0.5 MB)
    unsigned short* hb  = (unsigned short*)d_ws;
    unsigned short* w2t = hb + (size_t)B_ * H_;

    w2_transpose_bf16<<<(H_ * H_) / 256, 256, 0, stream>>>(W2, w2t);
    gather_bias_relu_bf16<<<B_ / 8, 256, 0, stream>>>(msg, W1, b1, hb);
    gemm_relu_wmma<<<dim3(B_ / BM, H_ / BN), 256, 0, stream>>>(hb, w2t, b2, out);
}
